// MultiHeadSelfAttention_31722628448459
// MI455X (gfx1250) — matrix operand order repaired and checked by reading, not compiled, not run on hardware
//
#include <hip/hip_runtime.h>
#include <hip/hip_bf16.h>

// ---- constants (match reference) ----
#define B_  2
#define S_  2048
#define D_  1024
#define H_  16
#define HD_ 64
#define M_  (B_ * S_)   // 4096 rows

typedef __attribute__((ext_vector_type(8)))  __bf16 v8bf;
typedef __attribute__((ext_vector_type(16))) __bf16 v16bf;
typedef __attribute__((ext_vector_type(8)))  float  v8f;

// ---------------------------------------------------------------------------
// Fragment loaders for V_WMMA_F32_16X16X32_BF16 (wave32 layouts, ISA 7.12.2)
// A 16x32 (MxK): lane L holds row M=L%16; K = (L<16?0:8)+{0..7} and +16..+23
// B 32x16 (KxN): lane L holds col N=L%16; K = (L<16?0:8)+{0..7} and +16..+23
// Both want K-contiguous memory (i.e. NT GEMM: A row-major, B^T row-major).
// ---------------------------------------------------------------------------
__device__ __forceinline__ v16bf load_frag_a(const __bf16* base, int stride, int lane) {
    const __bf16* p = base + (size_t)(lane & 15) * stride + ((lane & 16) >> 1);
    v8bf lo = *(const v8bf*)(p);
    v8bf hi = *(const v8bf*)(p + 16);
    return __builtin_shufflevector(lo, hi, 0,1,2,3,4,5,6,7,8,9,10,11,12,13,14,15);
}

__device__ __forceinline__ v16bf load_frag_b(const __bf16* base, int stride, int lane) {
    const __bf16* p = base + (size_t)(lane & 15) * stride + ((lane & 16) >> 1);
    v8bf lo = *(const v8bf*)(p);
    v8bf hi = *(const v8bf*)(p + 16);
    return __builtin_shufflevector(lo, hi, 0,1,2,3,4,5,6,7,8,9,10,11,12,13,14,15);
}

__device__ __forceinline__ v8f wmma_bf16(v16bf a, v16bf b, v8f c) {
    return __builtin_amdgcn_wmma_f32_16x16x32_bf16(false, a, false, b, (short)0, c, false, false);
}

// ---------------------------------------------------------------------------
// Kernel 0: f32 -> bf16 conversion
// ---------------------------------------------------------------------------
__global__ void cvt_bf16_kernel(const float* __restrict__ src, __bf16* __restrict__ dst, int n) {
    int i = blockIdx.x * blockDim.x + threadIdx.x;
    if (i < n) dst[i] = (__bf16)src[i];
}

// ---------------------------------------------------------------------------
// Kernel 1: Q projection  q[m,e] = sum_d x[m,d] * Wq[e,d]   (NT GEMM)
// Writes q bf16 row-major [M_, D_] and per-head transpose qT [B,H,HD,S].
// Wave tile: 32(M) x 64(N); K step 32.
// ---------------------------------------------------------------------------
__global__ void __launch_bounds__(128)
qproj_kernel(const __bf16* __restrict__ X, const __bf16* __restrict__ W,
             __bf16* __restrict__ Q, __bf16* __restrict__ QT) {
    const int lane = threadIdx.x & 31;
    const int wave = blockIdx.x * (blockDim.x >> 5) + (threadIdx.x >> 5);
    const int nt = wave & 15;          // 16 N tiles of 64
    const int mt = wave >> 4;          // 128 M tiles of 32
    const int m0 = mt * 32, n0 = nt * 64;

    v8f acc[2][4] = {};
    for (int k = 0; k < D_; k += 32) {
        v16bf a0 = load_frag_a(X + (size_t)m0 * D_ + k, D_, lane);
        v16bf a1 = load_frag_a(X + (size_t)(m0 + 16) * D_ + k, D_, lane);
        v16bf b0 = load_frag_b(W + (size_t)(n0 +  0) * D_ + k, D_, lane);
        v16bf b1 = load_frag_b(W + (size_t)(n0 + 16) * D_ + k, D_, lane);
        v16bf b2 = load_frag_b(W + (size_t)(n0 + 32) * D_ + k, D_, lane);
        v16bf b3 = load_frag_b(W + (size_t)(n0 + 48) * D_ + k, D_, lane);
        acc[0][0] = wmma_bf16(a0, b0, acc[0][0]);
        acc[0][1] = wmma_bf16(a0, b1, acc[0][1]);
        acc[0][2] = wmma_bf16(a0, b2, acc[0][2]);
        acc[0][3] = wmma_bf16(a0, b3, acc[0][3]);
        acc[1][0] = wmma_bf16(a1, b0, acc[1][0]);
        acc[1][1] = wmma_bf16(a1, b1, acc[1][1]);
        acc[1][2] = wmma_bf16(a1, b2, acc[1][2]);
        acc[1][3] = wmma_bf16(a1, b3, acc[1][3]);
    }
    // Epilogue: C/D layout lane L, vgpr r -> (M = r + (L<16?0:8), N = L%16)
    const int rl = (lane & 16) >> 1;
    const int cl = lane & 15;
#pragma unroll
    for (int i = 0; i < 2; ++i)
#pragma unroll
        for (int j = 0; j < 4; ++j)
#pragma unroll
            for (int r = 0; r < 8; ++r) {
                int m = m0 + i * 16 + rl + r;
                int n = n0 + j * 16 + cl;
                __bf16 v = (__bf16)acc[i][j][r];
                Q[(size_t)m * D_ + n] = v;
                int b = m >> 11, s = m & (S_ - 1);
                int h = n >> 6,  d = n & (HD_ - 1);
                QT[(((size_t)(b * H_ + h)) * HD_ + d) * S_ + s] = v;
            }
}

// ---------------------------------------------------------------------------
// Kernel 2: causal attention with online softmax (K = V = Q).
// One wave per (b, h, 16-row q tile). 32 k-columns per iteration.
// P tile is staged through LDS to convert D-fragment -> A-fragment layout.
// ---------------------------------------------------------------------------
__global__ void __launch_bounds__(128)
attn_kernel(const __bf16* __restrict__ Q, const __bf16* __restrict__ QT,
            __bf16* __restrict__ CTX) {
    __shared__ __bf16 plds[4][16 * 32];
    const int lane  = threadIdx.x & 31;
    const int wslot = threadIdx.x >> 5;
    const int wave  = blockIdx.x * 4 + wslot;      // 4096 waves total
    const int qt = wave & 127;                      // 128 q tiles
    const int h  = (wave >> 7) & 15;
    const int b  = wave >> 11;
    const int q0 = qt * 16;

    const __bf16* qbase  = Q  + (size_t)b * S_ * D_ + (size_t)h * HD_;      // + s*D_
    const __bf16* vtbase = QT + ((size_t)(b * H_ + h)) * HD_ * S_;          // + d*S_
    __bf16* lp = &plds[wslot][0];

    const int rl = (lane & 16) >> 1;
    const int cl = lane & 15;

    v8f ctx[4] = {};
    float mrow[8], lrow[8];
#pragma unroll
    for (int r = 0; r < 8; ++r) { mrow[r] = -1e30f; lrow[r] = 0.f; }
    const float sc = 0.125f * 1.44269504088896340736f;   // 1/sqrt(64) * log2(e)

    // Loop-invariant A fragments of the q tile (K = hd = 64 -> 2 fragments)
    v16bf aq0 = load_frag_a(qbase + (size_t)q0 * D_ +  0, D_, lane);
    v16bf aq1 = load_frag_a(qbase + (size_t)q0 * D_ + 32, D_, lane);

    for (int k0 = 0; k0 < q0 + 16; k0 += 32) {
        // ---- scores: two 16x16 subtiles, K=64 chained WMMAs ----
        v8f s[2];
#pragma unroll
        for (int t = 0; t < 2; ++t) {
            int kc = k0 + t * 16;
            v16bf bk0 = load_frag_b(qbase + (size_t)kc * D_ +  0, D_, lane);
            v16bf bk1 = load_frag_b(qbase + (size_t)kc * D_ + 32, D_, lane);
            v8f z = {};
            s[t] = wmma_bf16(aq1, bk1, wmma_bf16(aq0, bk0, z));
        }
        // ---- scale + causal mask (keep k <= q) ----
#pragma unroll
        for (int t = 0; t < 2; ++t) {
            int kc = k0 + t * 16 + cl;
#pragma unroll
            for (int r = 0; r < 8; ++r) {
                int rg = q0 + rl + r;
                float v = s[t][r] * sc;
                s[t][r] = (kc <= rg) ? v : -1e30f;
            }
        }
        // ---- row max across the 16 N-lanes ----
        float tmax[8];
#pragma unroll
        for (int r = 0; r < 8; ++r) tmax[r] = fmaxf(s[0][r], s[1][r]);
#pragma unroll
        for (int off = 1; off < 16; off <<= 1)
#pragma unroll
            for (int r = 0; r < 8; ++r)
                tmax[r] = fmaxf(tmax[r], __shfl_xor(tmax[r], off, 32));
        // ---- online softmax update ----
        float alpha[8], rsum[8];
#pragma unroll
        for (int r = 0; r < 8; ++r) {
            float mn = fmaxf(mrow[r], tmax[r]);
            alpha[r] = exp2f(mrow[r] - mn);
            mrow[r] = mn;
            rsum[r] = 0.f;
        }
#pragma unroll
        for (int t = 0; t < 2; ++t)
#pragma unroll
            for (int r = 0; r < 8; ++r) {
                float p = exp2f(s[t][r] - mrow[r]);
                s[t][r] = p;
                rsum[r] += p;
            }
#pragma unroll
        for (int off = 1; off < 16; off <<= 1)
#pragma unroll
            for (int r = 0; r < 8; ++r)
                rsum[r] += __shfl_xor(rsum[r], off, 32);
#pragma unroll
        for (int r = 0; r < 8; ++r) lrow[r] = lrow[r] * alpha[r] + rsum[r];
#pragma unroll
        for (int j = 0; j < 4; ++j)
#pragma unroll
            for (int r = 0; r < 8; ++r) ctx[j][r] *= alpha[r];

        // ---- D-fragment -> A-fragment via LDS (row-major 16x32 bf16) ----
#pragma unroll
        for (int t = 0; t < 2; ++t)
#pragma unroll
            for (int r = 0; r < 8; ++r)
                lp[(rl + r) * 32 + t * 16 + cl] = (__bf16)s[t][r];
        asm volatile("s_wait_dscnt 0" ::: "memory");
        v16bf pa = load_frag_a(lp, 32, lane);

        // ---- ctx += P(16x32) @ V(32x64), V^T from QT (K-contiguous) ----
#pragma unroll
        for (int j = 0; j < 4; ++j) {
            v16bf vb = load_frag_b(vtbase + (size_t)(j * 16) * S_ + k0, S_, lane);
            ctx[j] = wmma_bf16(pa, vb, ctx[j]);
        }
        asm volatile("s_wait_dscnt 0" ::: "memory");  // protect LDS WAR for next iter
    }

    // ---- normalize and store ctx tile (concat-heads layout [M_, D_]) ----
#pragma unroll
    for (int j = 0; j < 4; ++j)
#pragma unroll
        for (int r = 0; r < 8; ++r) {
            float v = ctx[j][r] / lrow[r];
            int row = q0 + rl + r;
            int col = h * HD_ + j * 16 + cl;
            CTX[((size_t)b * S_ + row) * D_ + col] = (__bf16)v;
        }
}

// ---------------------------------------------------------------------------
// Kernel 3: output projection  out[m,e] = sum_d ctx[m,d]*Wo[e,d] + bo[e]  (f32)
// ---------------------------------------------------------------------------
__global__ void __launch_bounds__(128)
oproj_kernel(const __bf16* __restrict__ X, const __bf16* __restrict__ W,
             const float* __restrict__ bias, float* __restrict__ OUT) {
    const int lane = threadIdx.x & 31;
    const int wave = blockIdx.x * (blockDim.x >> 5) + (threadIdx.x >> 5);
    const int nt = wave & 15;
    const int mt = wave >> 4;
    const int m0 = mt * 32, n0 = nt * 64;

    v8f acc[2][4] = {};
    for (int k = 0; k < D_; k += 32) {
        v16bf a0 = load_frag_a(X + (size_t)m0 * D_ + k, D_, lane);
        v16bf a1 = load_frag_a(X + (size_t)(m0 + 16) * D_ + k, D_, lane);
        v16bf b0 = load_frag_b(W + (size_t)(n0 +  0) * D_ + k, D_, lane);
        v16bf b1 = load_frag_b(W + (size_t)(n0 + 16) * D_ + k, D_, lane);
        v16bf b2 = load_frag_b(W + (size_t)(n0 + 32) * D_ + k, D_, lane);
        v16bf b3 = load_frag_b(W + (size_t)(n0 + 48) * D_ + k, D_, lane);
        acc[0][0] = wmma_bf16(a0, b0, acc[0][0]);
        acc[0][1] = wmma_bf16(a0, b1, acc[0][1]);
        acc[0][2] = wmma_bf16(a0, b2, acc[0][2]);
        acc[0][3] = wmma_bf16(a0, b3, acc[0][3]);
        acc[1][0] = wmma_bf16(a1, b0, acc[1][0]);
        acc[1][1] = wmma_bf16(a1, b1, acc[1][1]);
        acc[1][2] = wmma_bf16(a1, b2, acc[1][2]);
        acc[1][3] = wmma_bf16(a1, b3, acc[1][3]);
    }
    const int rl = (lane & 16) >> 1;
    const int cl = lane & 15;
#pragma unroll
    for (int i = 0; i < 2; ++i)
#pragma unroll
        for (int j = 0; j < 4; ++j) {
            int n = n0 + j * 16 + cl;
            float bv = bias[n];
#pragma unroll
            for (int r = 0; r < 8; ++r) {
                int m = m0 + i * 16 + rl + r;
                OUT[(size_t)m * D_ + n] = acc[i][j][r] + bv;
            }
        }
}

// ---------------------------------------------------------------------------
extern "C" void kernel_launch(void* const* d_in, const int* in_sizes, int n_in,
                              void* d_out, int out_size, void* d_ws, size_t ws_size,
                              hipStream_t stream) {
    const float* x  = (const float*)d_in[0];
    const float* Wq = (const float*)d_in[1];
    const float* Wo = (const float*)d_in[2];
    const float* bo = (const float*)d_in[3];

    char* ws = (char*)d_ws;
    const size_t NX = (size_t)M_ * D_;   // 4,194,304
    const size_t NW = (size_t)D_ * D_;   // 1,048,576
    __bf16* xb   = (__bf16*)(ws);
    __bf16* wqb  = (__bf16*)(ws + 2 * NX);
    __bf16* wob  = (__bf16*)(ws + 2 * NX + 2 * NW);
    __bf16* qb   = (__bf16*)(ws + 2 * NX + 4 * NW);
    __bf16* qtb  = (__bf16*)(ws + 4 * NX + 4 * NW);
    __bf16* ctxb = (__bf16*)(ws + 6 * NX + 4 * NW);

    cvt_bf16_kernel<<<(int)((NX + 255) / 256), 256, 0, stream>>>(x,  xb,  (int)NX);
    cvt_bf16_kernel<<<(int)((NW + 255) / 256), 256, 0, stream>>>(Wq, wqb, (int)NW);
    cvt_bf16_kernel<<<(int)((NW + 255) / 256), 256, 0, stream>>>(Wo, wob, (int)NW);

    // 2048 waves = 512 blocks x 4 waves
    qproj_kernel<<<512, 128, 0, stream>>>(xb, wqb, qb, qtb);
    // 4096 waves = 1024 blocks x 4 waves
    attn_kernel<<<1024, 128, 0, stream>>>(qb, qtb, ctxb);
    oproj_kernel<<<512, 128, 0, stream>>>(ctxb, wob, bo, (float*)d_out);
}
